// TransformerDecoderLayerBase_67018669686851
// MI455X (gfx1250) — matrix operand order repaired and checked by reading, not compiled, not run on hardware
//
#include <hip/hip_runtime.h>
#include <hip/hip_bf16.h>
#include <stdint.h>

// ---------------------------------------------------------------------------
// Problem constants (match reference: T,B,S,D,H,E,F,K = 1024,4,1024,1024,16,8,2048,2)
// ---------------------------------------------------------------------------
#define T_LEN   1024
#define B_SZ    4
#define S_LEN   1024
#define D_DIM   1024
#define H_HEADS 16
#define HD      64
#define E_EXP   8
#define F_DIM   2048
#define NTOK    (T_LEN * B_SZ)   // 4096 decoder tokens; encoder also 4096

typedef __bf16 bf16_t;
typedef __attribute__((ext_vector_type(16))) __bf16 bf16x16;
typedef __attribute__((ext_vector_type(8)))  float  floatx8;
typedef __attribute__((ext_vector_type(4)))  int    int4v;

// ---------------------------------------------------------------------------
// CDNA5 async global->LDS copy (GLOBAL_LOAD_ASYNC_TO_LDS_B128, ASYNCcnt).
// Builtin signature (from compiler diagnostic): (int4 as1*, int4 as3*, imm, imm)
// Guarded: falls back to register staging when the builtin is absent.
// ---------------------------------------------------------------------------
#if defined(__has_builtin)
#if __has_builtin(__builtin_amdgcn_global_load_async_to_lds_b128) && \
    __has_builtin(__builtin_amdgcn_s_wait_asynccnt)
#define USE_ASYNC_LDS 1
#endif
#endif
#ifndef USE_ASYNC_LDS
#define USE_ASYNC_LDS 0
#endif

#if USE_ASYNC_LDS
typedef __attribute__((address_space(1))) int4v gint4_t;
typedef __attribute__((address_space(3))) int4v lint4_t;
__device__ __forceinline__ void async_cp_b128(const bf16_t* g, bf16_t* l) {
  __builtin_amdgcn_global_load_async_to_lds_b128((gint4_t*)g, (lint4_t*)l, 0, 0);
}
__device__ __forceinline__ void async_wait_all() {
  __builtin_amdgcn_s_wait_asynccnt(0);
}
#else
__device__ __forceinline__ void async_cp_b128(const bf16_t* g, bf16_t* l) {
  *(uint4*)l = *(const uint4*)g;
}
__device__ __forceinline__ void async_wait_all() {}
#endif

// ---------------------------------------------------------------------------
// WMMA fragment loaders (wave32, 16-bit A 16x32 layout per CDNA5 ISA 7.12.2)
// A: lane m=lane&15, half=lane>>4; VGPR v holds K pair K0,K0+1 with
//    K0 = 2*(v&3) + 16*(v>>2) + 8*half
// B staged N-major in LDS (row = n, contiguous K), VGPR v pair at
//    K0 = 2*(v&3) + 16*(v>>2) + 8*half
// ---------------------------------------------------------------------------
__device__ __forceinline__ bf16x16 frag_a(const bf16_t* __restrict__ rowp, int half, int kofs) {
  union { bf16x16 v; uint32_t u[8]; } f;
#pragma unroll
  for (int vv = 0; vv < 8; vv++) {
    int k0 = kofs + 2 * (vv & 3) + 16 * (vv >> 2) + 8 * half;
    f.u[vv] = *(const uint32_t*)(rowp + k0);
  }
  return f.v;
}

__device__ __forceinline__ bf16x16 frag_b(const bf16_t* __restrict__ rowp, int half, int kofs) {
  union { bf16x16 v; uint32_t u[8]; } f;
#pragma unroll
  for (int vv = 0; vv < 8; vv++) {
    int k0 = kofs + 2 * (vv & 3) + 16 * (vv >> 2) + 8 * half;
    f.u[vv] = *(const uint32_t*)(rowp + k0);
  }
  return f.v;
}

// ---------------------------------------------------------------------------
// f32 -> bf16 conversion (weights once per launch, encoder activations)
// ---------------------------------------------------------------------------
__global__ void cvt_f32_to_bf16(const float* __restrict__ src, bf16_t* __restrict__ dst, int n) {
  int i = blockIdx.x * blockDim.x + threadIdx.x;
  int stride = gridDim.x * blockDim.x;
  for (; i < n; i += stride) dst[i] = (bf16_t)src[i];
}

// ---------------------------------------------------------------------------
// LayerNorm over D=1024, 256 threads / row, bf16 output for WMMA consumption
// ---------------------------------------------------------------------------
__global__ __launch_bounds__(256) void layernorm_bf16(
    const float* __restrict__ x, const float* __restrict__ g, const float* __restrict__ b,
    bf16_t* __restrict__ out) {
  const int row = blockIdx.x;
  const int tid = threadIdx.x;
  __shared__ float red[256];
  const float* xr = x + (size_t)row * D_DIM;
  float v[4];
  float s = 0.f;
#pragma unroll
  for (int i = 0; i < 4; i++) { v[i] = xr[tid + i * 256]; s += v[i]; }
  red[tid] = s; __syncthreads();
  for (int off = 128; off > 0; off >>= 1) {
    if (tid < off) red[tid] += red[tid + off];
    __syncthreads();
  }
  const float mu = red[0] / (float)D_DIM;
  __syncthreads();
  float s2 = 0.f;
#pragma unroll
  for (int i = 0; i < 4; i++) { float d = v[i] - mu; s2 += d * d; }
  red[tid] = s2; __syncthreads();
  for (int off = 128; off > 0; off >>= 1) {
    if (tid < off) red[tid] += red[tid + off];
    __syncthreads();
  }
  const float rstd = rsqrtf(red[0] / (float)D_DIM + 1e-5f);
#pragma unroll
  for (int i = 0; i < 4; i++) {
    int d = tid + i * 256;
    out[(size_t)row * D_DIM + d] = (bf16_t)((v[i] - mu) * rstd * g[d] + b[d]);
  }
}

// ---------------------------------------------------------------------------
// Tiled bf16 WMMA GEMM: C[M,N] = A[M,K] @ B[K,N]  (+fused epilogues)
// Block 256 thr = 8 waves (2x4), tile 128x128x32, double-buffered LDS with
// async global->LDS streaming of the A tile. M%128==N%128==K%32==0 required.
// ---------------------------------------------------------------------------
#define BM 128
#define BN 128
#define BK 32
#define LDK 40   // padded LDS row stride (elements) -> conflict-free frag reads

enum { GEMM_BF16_OUT = 0, GEMM_BF16_RELU = 1, GEMM_F32_RESID = 2, GEMM_F32_GATED_ACC = 3 };

__global__ __launch_bounds__(256) void gemm_bf16_kernel(
    const bf16_t* __restrict__ A, const bf16_t* __restrict__ Bw,
    const float* __restrict__ bias, int M, int N, int K,
    float alpha, int mode,
    bf16_t* __restrict__ outb, float* __restrict__ outf, float* __restrict__ outf2,
    const float* __restrict__ resid, const float* __restrict__ gates, int gate_col) {
  __shared__ __align__(16) bf16_t As[2][BM * LDK];
  __shared__ __align__(16) bf16_t Bs[2][BN * LDK];

  const int tid  = threadIdx.x;
  const int lane = tid & 31;
  const int wave = tid >> 5;
  const int wm = wave >> 2;       // 0..1 (64 rows each)
  const int wn = wave & 3;        // 0..3 (32 cols each)
  const int half = lane >> 4, l16 = lane & 15;
  const int m0 = blockIdx.y * BM, n0 = blockIdx.x * BN;

  floatx8 acc[4][2];
#pragma unroll
  for (int fm = 0; fm < 4; fm++)
#pragma unroll
    for (int fn = 0; fn < 2; fn++) acc[fm][fn] = (floatx8)0.0f;

  // per-iteration staging: A tile streamed straight to LDS (async path),
  // B tile fetched to registers then transpose-scattered to N-major LDS.
  auto stage_A = [&](int bufi, int kt) {
#pragma unroll
    for (int i = 0; i < 2; i++) {
      int c = i * 256 + tid;                 // 512 chunks of 8 bf16
      int row = c >> 2, kk = (c & 3) * 8;
      async_cp_b128(A + (size_t)(m0 + row) * K + kt + kk, &As[bufi][row * LDK + kk]);
    }
  };
  auto load_B = [&](uint4* breg, int kt) {
#pragma unroll
    for (int i = 0; i < 2; i++) {
      int c = i * 256 + tid;                 // 512 chunks of 8 bf16
      int kr = c >> 4, nn = (c & 15) * 8;
      breg[i] = *(const uint4*)(Bw + (size_t)(kt + kr) * N + n0 + nn);
    }
  };
  auto store_B = [&](int bufi, const uint4* breg) {
#pragma unroll
    for (int i = 0; i < 2; i++) {
      int c = i * 256 + tid;
      int kr = c >> 4, nn = (c & 15) * 8;
      union { uint4 u; bf16_t h[8]; } d;
      d.u = breg[i];
#pragma unroll
      for (int j = 0; j < 8; j++) Bs[bufi][(nn + j) * LDK + kr] = d.h[j];
    }
  };

  // prologue: stage tile 0
  uint4 breg[2];
  stage_A(0, 0);
  load_B(breg, 0);
  store_B(0, breg);
  async_wait_all();
  __syncthreads();

  int buf = 0;
  for (int kt = 0; kt < K; kt += BK) {
    const int nxt = buf ^ 1;
    const bool more = (kt + BK) < K;
    if (more) {
      load_B(breg, kt + BK);       // global loads overlap the WMMAs below
      stage_A(nxt, kt + BK);       // async stream into the other buffer
    }

    bf16x16 af[4], bfr[2];
#pragma unroll
    for (int fm = 0; fm < 4; fm++)
      af[fm] = frag_a(&As[buf][(wm * 64 + fm * 16 + l16) * LDK], half, 0);
#pragma unroll
    for (int fn = 0; fn < 2; fn++)
      bfr[fn] = frag_b(&Bs[buf][(wn * 32 + fn * 16 + l16) * LDK], half, 0);
#pragma unroll
    for (int fm = 0; fm < 4; fm++)
#pragma unroll
      for (int fn = 0; fn < 2; fn++)
        acc[fm][fn] = __builtin_amdgcn_wmma_f32_16x16x32_bf16(
            false, af[fm], false, bfr[fn], (short)0, acc[fm][fn], false, false);

    if (more) {
      store_B(nxt, breg);
      async_wait_all();
    }
    __syncthreads();
    buf = nxt;
  }

  // ---- fused epilogue ----
#pragma unroll
  for (int fm = 0; fm < 4; fm++)
#pragma unroll
    for (int fn = 0; fn < 2; fn++) {
      const int col = n0 + wn * 32 + fn * 16 + l16;
      const float bv = bias ? bias[col] : 0.0f;
#pragma unroll
      for (int r = 0; r < 8; r++) {
        const int row = m0 + wm * 64 + fm * 16 + half * 8 + r;
        const size_t idx = (size_t)row * N + col;
        const float v = acc[fm][fn][r] + bv;
        if (mode == GEMM_BF16_OUT) {
          outb[idx] = (bf16_t)(v * alpha);
        } else if (mode == GEMM_BF16_RELU) {
          outb[idx] = (bf16_t)fmaxf(v, 0.0f);
        } else if (mode == GEMM_F32_RESID) {
          const float o = resid[idx] + v;
          outf[idx] = o;
          if (outf2) outf2[idx] = o;
        } else { // GEMM_F32_GATED_ACC
          const float gg = gates[(size_t)row * E_EXP + gate_col];
          if (gg != 0.0f) outf[idx] += gg * v;
        }
      }
    }
}

// ---------------------------------------------------------------------------
// Flash-style attention: per (b,h) and 64-query tile; online softmax in LDS.
// q already scaled by 1/sqrt(hd) at projection. Tokens addressed as [L,B,D].
// Block = 128 threads = 4 waves; wave w owns query rows w*16..w*16+15.
// ---------------------------------------------------------------------------
__global__ __launch_bounds__(128) void attn_kernel(
    const bf16_t* __restrict__ q, const bf16_t* __restrict__ k, const bf16_t* __restrict__ v,
    bf16_t* __restrict__ o, int Lk, int causal) {
  const int q0 = blockIdx.x * 64;
  const int bh = blockIdx.y;
  const int b = bh / H_HEADS, h = bh % H_HEADS;
  const int tid = threadIdx.x;
  const int lane = tid & 31, wave = tid >> 5;
  const int half = lane >> 4, l16 = lane & 15;
  const size_t TOKSTRIDE = (size_t)B_SZ * D_DIM;

  __shared__ __align__(16) bf16_t Qs[64 * 72];
  __shared__ __align__(16) bf16_t Ks[64 * 72];
  __shared__ __align__(16) bf16_t Vt[64 * 72];   // Vt[d][key]
  __shared__ float Ss[64 * 68];
  __shared__ float rowM[64], rowL[64], rowF[64];

  // stage Q tile (64 rows x 64 dims) via async global->LDS
  const size_t qbase = ((size_t)q0 * B_SZ + b) * D_DIM + h * HD;
#pragma unroll
  for (int i = 0; i < 4; i++) {
    int c = i * 128 + tid;
    int r = c >> 3, dd = (c & 7) * 8;
    async_cp_b128(q + qbase + (size_t)r * TOKSTRIDE + dd, &Qs[r * 72 + dd]);
  }
  if (tid < 64) { rowM[tid] = -3.0e38f; rowL[tid] = 0.0f; }
  async_wait_all();
  __syncthreads();

  bf16x16 qf[2];
#pragma unroll
  for (int ks = 0; ks < 2; ks++)
    qf[ks] = frag_a(&Qs[(wave * 16 + l16) * 72], half, ks * 32);

  floatx8 of[4];
#pragma unroll
  for (int fn = 0; fn < 4; fn++) of[fn] = (floatx8)0.0f;

  const int ntiles = causal ? (q0 / 64 + 1) : (Lk / 64);
  for (int jt = 0; jt < ntiles; jt++) {
    const int j0 = jt * 64;
    const size_t kbase = ((size_t)j0 * B_SZ + b) * D_DIM + h * HD;
    // K tile: async straight to LDS; V tile: batched loads then transpose scatter
    uint4 vreg[4];
#pragma unroll
    for (int i = 0; i < 4; i++) {
      int c = i * 128 + tid;
      int r = c >> 3, dd = (c & 7) * 8;
      async_cp_b128(k + kbase + (size_t)r * TOKSTRIDE + dd, &Ks[r * 72 + dd]);
      vreg[i] = *(const uint4*)(v + kbase + (size_t)r * TOKSTRIDE + dd);
    }
#pragma unroll
    for (int i = 0; i < 4; i++) {
      int c = i * 128 + tid;
      int r = c >> 3, dd = (c & 7) * 8;
      union { uint4 u; bf16_t hh[8]; } dv;
      dv.u = vreg[i];
#pragma unroll
      for (int j = 0; j < 8; j++) Vt[(dd + j) * 72 + r] = dv.hh[j];
    }
    async_wait_all();
    __syncthreads();

    // S = Q @ K^T  (16 q-rows x 64 keys per wave)
    floatx8 sacc[4];
#pragma unroll
    for (int fn = 0; fn < 4; fn++) sacc[fn] = (floatx8)0.0f;
#pragma unroll
    for (int ks = 0; ks < 2; ks++)
#pragma unroll
      for (int fn = 0; fn < 4; fn++) {
        bf16x16 kf = frag_b(&Ks[(fn * 16 + l16) * 72], half, ks * 32);
        sacc[fn] = __builtin_amdgcn_wmma_f32_16x16x32_bf16(
            false, qf[ks], false, kf, (short)0, sacc[fn], false, false);
      }
#pragma unroll
    for (int fn = 0; fn < 4; fn++)
#pragma unroll
      for (int r = 0; r < 8; r++)
        Ss[(wave * 16 + half * 8 + r) * 68 + fn * 16 + l16] = sacc[fn][r];
    __syncthreads();

    // online softmax per query row (threads 0..63 own one row each)
    if (tid < 64) {
      const int row = tid;
      const int qglob = q0 + row;
      const float mold = rowM[row];
      float mx = mold;
      for (int j = 0; j < 64; j++) {
        float s = Ss[row * 68 + j];
        if (causal && (j0 + j) > qglob) s = -1.0e30f;
        Ss[row * 68 + j] = s;
        mx = fmaxf(mx, s);
      }
      const float f = __expf(mold - mx);
      float sum = 0.f;
      for (int j = 0; j < 64; j++) {
        float p = __expf(Ss[row * 68 + j] - mx);
        Ss[row * 68 + j] = p;
        sum += p;
      }
      rowM[row] = mx;
      rowL[row] = rowL[row] * f + sum;
      rowF[row] = f;
    }
    __syncthreads();

    // rescale O accumulators by per-row factor
    float fr[8];
#pragma unroll
    for (int r = 0; r < 8; r++) fr[r] = rowF[wave * 16 + half * 8 + r];
#pragma unroll
    for (int fn = 0; fn < 4; fn++)
#pragma unroll
      for (int r = 0; r < 8; r++) of[fn][r] *= fr[r];

    // O += P @ V
#pragma unroll
    for (int ks = 0; ks < 2; ks++) {
      union { bf16x16 v; uint32_t u[8]; } pf;
      const int prow = wave * 16 + l16;
#pragma unroll
      for (int vv = 0; vv < 8; vv++) {
        int k0 = ks * 32 + 2 * (vv & 3) + 16 * (vv >> 2) + 8 * half;
        union { uint32_t u; bf16_t h[2]; } pk;
        pk.h[0] = (bf16_t)Ss[prow * 68 + k0];
        pk.h[1] = (bf16_t)Ss[prow * 68 + k0 + 1];
        pf.u[vv] = pk.u;
      }
#pragma unroll
      for (int fn = 0; fn < 4; fn++) {
        bf16x16 vf = frag_b(&Vt[(fn * 16 + l16) * 72], half, ks * 32);
        of[fn] = __builtin_amdgcn_wmma_f32_16x16x32_bf16(
            false, pf.v, false, vf, (short)0, of[fn], false, false);
      }
    }
    __syncthreads();
  }

  // normalize and store (bf16, same [L,B,D] head-sliced layout)
  float li[8];
#pragma unroll
  for (int r = 0; r < 8; r++) li[r] = rowL[wave * 16 + half * 8 + r];
#pragma unroll
  for (int fn = 0; fn < 4; fn++)
#pragma unroll
    for (int r = 0; r < 8; r++) {
      const int rowg = q0 + wave * 16 + half * 8 + r;
      const int col = fn * 16 + l16;
      o[((size_t)rowg * B_SZ + b) * D_DIM + h * HD + col] = (bf16_t)(of[fn][r] / li[r]);
    }
}

// ---------------------------------------------------------------------------
// MoE gating: logits = xn @ gate_W (D x 8), top-2 softmax -> sparse gates[N,8]
// ---------------------------------------------------------------------------
__global__ __launch_bounds__(64) void gate_topk_kernel(
    const bf16_t* __restrict__ xn, const float* __restrict__ gW, float* __restrict__ gates) {
  const int row = blockIdx.x, tid = threadIdx.x;
  float acc[E_EXP];
#pragma unroll
  for (int e = 0; e < E_EXP; e++) acc[e] = 0.f;
  for (int d = tid; d < D_DIM; d += 64) {
    const float xv = (float)xn[(size_t)row * D_DIM + d];
    const float* g = gW + (size_t)d * E_EXP;
#pragma unroll
    for (int e = 0; e < E_EXP; e++) acc[e] += xv * g[e];
  }
  __shared__ float red[64][E_EXP];
#pragma unroll
  for (int e = 0; e < E_EXP; e++) red[tid][e] = acc[e];
  __syncthreads();
  for (int off = 32; off > 0; off >>= 1) {
    if (tid < off)
#pragma unroll
      for (int e = 0; e < E_EXP; e++) red[tid][e] += red[tid + off][e];
    __syncthreads();
  }
  if (tid == 0) {
    float v1 = -3.0e38f, v2 = -3.0e38f; int i1 = 0, i2 = 0;
#pragma unroll
    for (int e = 0; e < E_EXP; e++) {
      const float lv = red[0][e];
      if (lv > v1) { v2 = v1; i2 = i1; v1 = lv; i1 = e; }
      else if (lv > v2) { v2 = lv; i2 = e; }
    }
    const float e2 = __expf(v2 - v1);
    const float g1 = 1.0f / (1.0f + e2);
    const float g2 = e2 / (1.0f + e2);
#pragma unroll
    for (int e = 0; e < E_EXP; e++) gates[(size_t)row * E_EXP + e] = 0.0f;
    gates[(size_t)row * E_EXP + i1] = g1;
    gates[(size_t)row * E_EXP + i2] = g2;
  }
}

// ---------------------------------------------------------------------------
// Host orchestration
// ---------------------------------------------------------------------------
extern "C" void kernel_launch(void* const* d_in, const int* in_sizes, int n_in,
                              void* d_out, int out_size, void* d_ws, size_t ws_size,
                              hipStream_t stream) {
  (void)in_sizes; (void)n_in; (void)out_size; (void)ws_size;
  const float* x        = (const float*)d_in[0];
  const float* enc      = (const float*)d_in[1];
  // d_in[2]: causal mask, applied analytically in attn_kernel
  const float* ln1_g = (const float*)d_in[3];  const float* ln1_b = (const float*)d_in[4];
  const float* ln2_g = (const float*)d_in[5];  const float* ln2_b = (const float*)d_in[6];
  const float* ln3_g = (const float*)d_in[7];  const float* ln3_b = (const float*)d_in[8];
  const float* sa_Wq = (const float*)d_in[9];  const float* sa_bq = (const float*)d_in[10];
  const float* sa_Wk = (const float*)d_in[11]; const float* sa_bk = (const float*)d_in[12];
  const float* sa_Wv = (const float*)d_in[13]; const float* sa_bv = (const float*)d_in[14];
  const float* sa_Wo = (const float*)d_in[15]; const float* sa_bo = (const float*)d_in[16];
  const float* ca_Wq = (const float*)d_in[17]; const float* ca_bq = (const float*)d_in[18];
  const float* ca_Wk = (const float*)d_in[19]; const float* ca_bk = (const float*)d_in[20];
  const float* ca_Wv = (const float*)d_in[21]; const float* ca_bv = (const float*)d_in[22];
  const float* ca_Wo = (const float*)d_in[23]; const float* ca_bo = (const float*)d_in[24];
  const float* gate_W = (const float*)d_in[25];
  const float* moe_W1 = (const float*)d_in[26]; const float* moe_b1 = (const float*)d_in[27];
  const float* moe_W2 = (const float*)d_in[28]; const float* moe_b2 = (const float*)d_in[29];
  float* outp = (float*)d_out;

  char* ws = (char*)d_ws;
  size_t off = 0;
  auto alloc = [&](size_t bytes) -> void* {
    void* p = ws + off;
    off = (off + bytes + 255) & ~(size_t)255;
    return p;
  };

  const size_t DD = (size_t)D_DIM * D_DIM;
  bf16_t* w_saq = (bf16_t*)alloc(DD * 2);
  bf16_t* w_sak = (bf16_t*)alloc(DD * 2);
  bf16_t* w_sav = (bf16_t*)alloc(DD * 2);
  bf16_t* w_sao = (bf16_t*)alloc(DD * 2);
  bf16_t* w_caq = (bf16_t*)alloc(DD * 2);
  bf16_t* w_cak = (bf16_t*)alloc(DD * 2);
  bf16_t* w_cav = (bf16_t*)alloc(DD * 2);
  bf16_t* w_cao = (bf16_t*)alloc(DD * 2);
  bf16_t* w_m1  = (bf16_t*)alloc((size_t)E_EXP * D_DIM * F_DIM * 2);
  bf16_t* w_m2  = (bf16_t*)alloc((size_t)E_EXP * F_DIM * D_DIM * 2);
  bf16_t* enc_b = (bf16_t*)alloc((size_t)NTOK * D_DIM * 2);
  bf16_t* xn    = (bf16_t*)alloc((size_t)NTOK * D_DIM * 2);
  bf16_t* qb    = (bf16_t*)alloc((size_t)NTOK * D_DIM * 2);
  bf16_t* kb    = (bf16_t*)alloc((size_t)NTOK * D_DIM * 2);
  bf16_t* vb    = (bf16_t*)alloc((size_t)NTOK * D_DIM * 2);
  bf16_t* atn   = (bf16_t*)alloc((size_t)NTOK * D_DIM * 2);
  bf16_t* hbuf  = (bf16_t*)alloc((size_t)NTOK * F_DIM * 2);
  float*  x1    = (float*)alloc((size_t)NTOK * D_DIM * 4);
  float*  x2    = (float*)alloc((size_t)NTOK * D_DIM * 4);
  float*  gates = (float*)alloc((size_t)NTOK * E_EXP * 4);

  auto cvt = [&](const float* s, bf16_t* d, size_t n) {
    cvt_f32_to_bf16<<<dim3(2048), dim3(256), 0, stream>>>(s, d, (int)n);
  };
  auto gemm = [&](const bf16_t* A, const bf16_t* Bw, const float* bias,
                  int M, int N, int K, float alpha, int mode,
                  bf16_t* ob, float* of, float* of2, const float* resid,
                  const float* gg, int gcol) {
    dim3 grid(N / BN, M / BM);
    gemm_bf16_kernel<<<grid, dim3(256), 0, stream>>>(A, Bw, bias, M, N, K, alpha, mode,
                                                     ob, of, of2, resid, gg, gcol);
  };

  // ---- weight / encoder conversion to bf16 ----
  cvt(sa_Wq, w_saq, DD); cvt(sa_Wk, w_sak, DD); cvt(sa_Wv, w_sav, DD); cvt(sa_Wo, w_sao, DD);
  cvt(ca_Wq, w_caq, DD); cvt(ca_Wk, w_cak, DD); cvt(ca_Wv, w_cav, DD); cvt(ca_Wo, w_cao, DD);
  cvt(moe_W1, w_m1, (size_t)E_EXP * D_DIM * F_DIM);
  cvt(moe_W2, w_m2, (size_t)E_EXP * F_DIM * D_DIM);
  cvt(enc, enc_b, (size_t)NTOK * D_DIM);

  const float qscale = 0.125f;  // 1/sqrt(64)

  // ---- self-attention (pre-LN) ----
  layernorm_bf16<<<dim3(NTOK), dim3(256), 0, stream>>>(x, ln1_g, ln1_b, xn);
  gemm(xn, w_saq, sa_bq, NTOK, D_DIM, D_DIM, qscale, GEMM_BF16_OUT, qb, nullptr, nullptr, nullptr, nullptr, 0);
  gemm(xn, w_sak, sa_bk, NTOK, D_DIM, D_DIM, 1.0f,  GEMM_BF16_OUT, kb, nullptr, nullptr, nullptr, nullptr, 0);
  gemm(xn, w_sav, sa_bv, NTOK, D_DIM, D_DIM, 1.0f,  GEMM_BF16_OUT, vb, nullptr, nullptr, nullptr, nullptr, 0);
  attn_kernel<<<dim3(T_LEN / 64, B_SZ * H_HEADS), dim3(128), 0, stream>>>(qb, kb, vb, atn, T_LEN, 1);
  gemm(atn, w_sao, sa_bo, NTOK, D_DIM, D_DIM, 1.0f, GEMM_F32_RESID, nullptr, x1, nullptr, x, nullptr, 0);

  // ---- cross-attention (pre-LN) ----
  layernorm_bf16<<<dim3(NTOK), dim3(256), 0, stream>>>(x1, ln2_g, ln2_b, xn);
  gemm(xn,    w_caq, ca_bq, NTOK, D_DIM, D_DIM, qscale, GEMM_BF16_OUT, qb, nullptr, nullptr, nullptr, nullptr, 0);
  gemm(enc_b, w_cak, ca_bk, NTOK, D_DIM, D_DIM, 1.0f,  GEMM_BF16_OUT, kb, nullptr, nullptr, nullptr, nullptr, 0);
  gemm(enc_b, w_cav, ca_bv, NTOK, D_DIM, D_DIM, 1.0f,  GEMM_BF16_OUT, vb, nullptr, nullptr, nullptr, nullptr, 0);
  attn_kernel<<<dim3(T_LEN / 64, B_SZ * H_HEADS), dim3(128), 0, stream>>>(qb, kb, vb, atn, S_LEN, 0);
  // writes x2 (for LN3) AND initializes d_out with the residual stream
  gemm(atn, w_cao, ca_bo, NTOK, D_DIM, D_DIM, 1.0f, GEMM_F32_RESID, nullptr, x2, outp, x1, nullptr, 0);

  // ---- MoE FFN (pre-LN, dense gate-weighted expert sum, top-2 softmax gates) ----
  layernorm_bf16<<<dim3(NTOK), dim3(256), 0, stream>>>(x2, ln3_g, ln3_b, xn);
  gate_topk_kernel<<<dim3(NTOK), dim3(64), 0, stream>>>(xn, gate_W, gates);
  for (int e = 0; e < E_EXP; e++) {
    const bf16_t* W1e = w_m1 + (size_t)e * D_DIM * F_DIM;
    const bf16_t* W2e = w_m2 + (size_t)e * F_DIM * D_DIM;
    const float* b1e = moe_b1 + (size_t)e * F_DIM;
    const float* b2e = moe_b2 + (size_t)e * D_DIM;
    gemm(xn, W1e, b1e, NTOK, F_DIM, D_DIM, 1.0f, GEMM_BF16_RELU, hbuf, nullptr, nullptr, nullptr, nullptr, 0);
    gemm(hbuf, W2e, b2e, NTOK, D_DIM, F_DIM, 1.0f, GEMM_F32_GATED_ACC, nullptr, outp, nullptr, nullptr, gates, e);
  }
}
